// DeepSeekPhiSurreal_56650618634412
// MI455X (gfx1250) — matrix operand order repaired and checked by reading, not compiled, not run on hardware
//
#include <hip/hip_runtime.h>
#include <hip/hip_bf16.h>

// ---------------------------------------------------------------------------
// DeepSeekPhiSurreal on MI455X (gfx1250): bf16 WMMA GEMMs + chunked EMA scan
// ---------------------------------------------------------------------------

#define T_SEQ   4096
#define DIM_D   1910
#define DIM_DP  1920   // 60 * 32, K padded for 16x16x32 bf16 WMMA
#define NLAYERS 12
#define VOCAB   10000

#define ALPHA_F 0.61803398874989484820f   // 1/phi
#define BETA_F  0.38196601125010515180f   // 1/phi^2 == 1 - 1/phi

typedef __attribute__((ext_vector_type(16))) __bf16 v16bf;
typedef __attribute__((ext_vector_type(8)))  __bf16 v8bf;
typedef __attribute__((ext_vector_type(8)))  float  v8f;

// A-fragment (16-bit A 16x32, ISA 7.12.2): per lane, halves 0..7 = K[k0+off..+7],
// halves 8..15 = K[k0+off+16..+23] of row (lane&15); off = (lane>=16) ? 8 : 0.
__device__ __forceinline__ v16bf load_a_frag(const __bf16* p) {
    v8bf lo = *(const v8bf*)p;
    v8bf hi = *(const v8bf*)(p + 16);
    return __builtin_shufflevector(lo, hi, 0,1,2,3,4,5,6,7,8,9,10,11,12,13,14,15);
}

#define WMMA_BF16(a, b, c) __builtin_amdgcn_wmma_f32_16x16x32_bf16( \
    false, (a), false, (b), (short)0, (c), false, false)

// ---------------------------------------------------------------------------
// Embedding gather: h[t,d] = embed[token_ids[t], d]
// ---------------------------------------------------------------------------
__global__ void gather_embed(const int* __restrict__ tok,
                             const float* __restrict__ emb,
                             float* __restrict__ h)
{
    size_t idx = (size_t)blockIdx.x * 256 + threadIdx.x;
    if (idx >= (size_t)T_SEQ * DIM_D) return;
    int t = (int)(idx / DIM_D);
    int d = (int)(idx % DIM_D);
    h[idx] = emb[(size_t)tok[t] * DIM_D + d];
}

// ---------------------------------------------------------------------------
// f32 -> bf16 with zero padding to (prows, pcols)
// ---------------------------------------------------------------------------
__global__ void convert_pad_bf16(const float* __restrict__ src,
                                 __hip_bfloat16* __restrict__ dst,
                                 int rows, int cols, int prows, int pcols)
{
    size_t idx = (size_t)blockIdx.x * 256 + threadIdx.x;
    size_t total = (size_t)prows * pcols;
    if (idx >= total) return;
    int r = (int)(idx / pcols);
    int c = (int)(idx % pcols);
    float v = (r < rows && c < cols) ? src[(size_t)r * cols + c] : 0.0f;
    dst[idx] = __float2bfloat16(v);
}

// ---------------------------------------------------------------------------
// retro_attn + residual:  y = x + alpha*fwdEMA + beta*bwd7tap
// Chunked over T: 64-token warmup re-derives the EMA state exactly to f32
// precision ((1-alpha)^64 ~ 2e-27).
// ---------------------------------------------------------------------------
__global__ __launch_bounds__(256)
void attn_resid(const float* __restrict__ h, float* __restrict__ y)
{
    int d = blockIdx.x * 256 + threadIdx.x;
    if (d >= DIM_D) return;

    constexpr int CH = 128, WARM = 64;
    constexpr float A   = ALPHA_F;
    constexpr float OMA = BETA_F;          // 1 - alpha
    constexpr float W0 = A;                              // on x_{t-6}
    constexpr float W1 = A * OMA;
    constexpr float W2 = A * OMA * OMA;
    constexpr float W3 = A * OMA * OMA * OMA;
    constexpr float W4 = A * OMA * OMA * OMA * OMA;
    constexpr float W5 = A * OMA * OMA * OMA * OMA * OMA;
    constexpr float W6 = OMA * OMA * OMA * OMA * OMA * OMA;  // on x_t

    int tstart = blockIdx.y * CH;
    int tend   = tstart + CH; if (tend > T_SEQ) tend = T_SEQ;
    int t0     = tstart - WARM; if (t0 < 0) t0 = 0;

    float f = 0.0f;
    float r0=0,r1=0,r2=0,r3=0,r4=0,r5=0,r6=0;   // x_{t-6} .. x_t

    for (int t = t0; t < tend; ++t) {
        float x = h[(size_t)t * DIM_D + d];
        f = (t == 0) ? x : (A * x + OMA * f);
        r0=r1; r1=r2; r2=r3; r3=r4; r4=r5; r5=r6; r6=x;
        if (t >= tstart) {
            float bwd = (t < 6) ? x
                : (W0*r0 + W1*r1 + W2*r2 + W3*r3 + W4*r4 + W5*r5 + W6*r6);
            y[(size_t)t * DIM_D + d] = x + A * f + BETA_F * bwd;
        }
    }
}

// ---------------------------------------------------------------------------
// Row layernorm of (a*X + b*Y); optional f32 out (stride DIM_D) and
// zero-padded bf16 out (stride DIM_DP). One block per row.
// ---------------------------------------------------------------------------
__global__ __launch_bounds__(256)
void ln_fused(const float* __restrict__ X, const float* __restrict__ Y,
              float a, float b,
              const float* __restrict__ g, const float* __restrict__ bb,
              float* __restrict__ o32, __hip_bfloat16* __restrict__ o16,
              int ldY)
{
    int row = blockIdx.x;
    const float* xr = X + (size_t)row * DIM_D;
    const float* yr = Y ? (Y + (size_t)row * ldY) : nullptr;

    __shared__ float s1[256], s2[256];
    float sum = 0.0f, sq = 0.0f;
    for (int d = threadIdx.x; d < DIM_D; d += 256) {
        float v = a * xr[d];
        if (yr) v += b * yr[d];
        sum += v; sq += v * v;
    }
    s1[threadIdx.x] = sum; s2[threadIdx.x] = sq;
    __syncthreads();
    for (int s = 128; s > 0; s >>= 1) {
        if (threadIdx.x < s) {
            s1[threadIdx.x] += s1[threadIdx.x + s];
            s2[threadIdx.x] += s2[threadIdx.x + s];
        }
        __syncthreads();
    }
    float mu   = s1[0] * (1.0f / DIM_D);
    float var  = s2[0] * (1.0f / DIM_D) - mu * mu;
    float rstd = rsqrtf(var + 1e-5f);

    for (int d = threadIdx.x; d < DIM_DP; d += 256) {
        float o = 0.0f;
        if (d < DIM_D) {
            float v = a * xr[d];
            if (yr) v += b * yr[d];
            o = (v - mu) * rstd * g[d] + bb[d];
            if (o32) o32[(size_t)row * DIM_D + d] = o;
        }
        if (o16) o16[(size_t)row * DIM_DP + d] = __float2bfloat16(o);
    }
}

// ---------------------------------------------------------------------------
// C[M,N] = A[M,K] * B[N,K]^T  in bf16 with f32 WMMA accumulation.
// Block = 256 threads (8 wave32), block tile 128x128, wave tile 32x64
// (2x4 grid of 16x16x32 WMMAs), software-pipelined fragment loads.
//
// Out-of-range B columns (final GEMM, N=10000) are handled by CLAMPING the
// row pointer to row 0: WMMA output column c depends only on B-fragment
// lane c, so clamped-garbage columns are exactly the ones the store guard
// discards. This keeps the inner loop branch-free and exec-mask-free.
// ---------------------------------------------------------------------------
__global__ __launch_bounds__(256)
void wmma_gemm_nt_bf16(const __hip_bfloat16* __restrict__ A_,
                       const __hip_bfloat16* __restrict__ B_,
                       float* __restrict__ C,
                       int N, int K, int ldc)
{
    const __bf16* A = reinterpret_cast<const __bf16*>(A_);
    const __bf16* B = reinterpret_cast<const __bf16*>(B_);

    const int lane = threadIdx.x & 31;
    const int wave = threadIdx.x >> 5;
    const int wm   = wave >> 1;           // 0..3  -> M offset wm*32
    const int wn   = wave & 1;            // 0..1  -> N offset wn*64
    const int half = lane >> 4;           // 0/1
    const int l16  = lane & 15;

    const int rowBase = blockIdx.y * 128 + wm * 32;
    const int colBase = blockIdx.x * 128 + wn * 64;

    const int aoff = half * 8;            // A half-K offset (elements)
    const int boff = half * 8;            // B half-K offset (elements)

    v8f acc[2][4] = {};

    const __bf16* a0p = A + (size_t)(rowBase + l16) * K + aoff;
    const __bf16* a1p = a0p + (size_t)16 * K;

    const __bf16* bp[4];
#pragma unroll
    for (int j = 0; j < 4; ++j) {
        int n = colBase + j * 16 + l16;
        bp[j] = B + (size_t)((n < N) ? n : 0) * K + boff;   // clamp: safe load
    }

    // ---- software pipeline: prologue ----
    v16bf a0 = load_a_frag(a0p);
    v16bf a1 = load_a_frag(a1p);
    v16bf b0 = load_a_frag(bp[0]);
    v16bf b1 = load_a_frag(bp[1]);
    v16bf b2 = load_a_frag(bp[2]);
    v16bf b3 = load_a_frag(bp[3]);

    for (int k0 = 32; k0 < K; k0 += 32) {
        // unconditional speculative stream-ahead on A (HBM side)
        __builtin_prefetch(a0p + k0 + 32, 0, 1);
        __builtin_prefetch(a1p + k0 + 32, 0, 1);

        // next iteration's fragments (overlap with this iteration's WMMAs)
        v16bf a0n = load_a_frag(a0p + k0);
        v16bf a1n = load_a_frag(a1p + k0);
        v16bf b0n = load_a_frag(bp[0] + k0);
        v16bf b1n = load_a_frag(bp[1] + k0);
        v16bf b2n = load_a_frag(bp[2] + k0);
        v16bf b3n = load_a_frag(bp[3] + k0);

        acc[0][0] = WMMA_BF16(a0, b0, acc[0][0]);
        acc[1][0] = WMMA_BF16(a1, b0, acc[1][0]);
        acc[0][1] = WMMA_BF16(a0, b1, acc[0][1]);
        acc[1][1] = WMMA_BF16(a1, b1, acc[1][1]);
        acc[0][2] = WMMA_BF16(a0, b2, acc[0][2]);
        acc[1][2] = WMMA_BF16(a1, b2, acc[1][2]);
        acc[0][3] = WMMA_BF16(a0, b3, acc[0][3]);
        acc[1][3] = WMMA_BF16(a1, b3, acc[1][3]);

        a0 = a0n; a1 = a1n;
        b0 = b0n; b1 = b1n; b2 = b2n; b3 = b3n;
    }

    // ---- epilogue: last k-step ----
    acc[0][0] = WMMA_BF16(a0, b0, acc[0][0]);
    acc[1][0] = WMMA_BF16(a1, b0, acc[1][0]);
    acc[0][1] = WMMA_BF16(a0, b1, acc[0][1]);
    acc[1][1] = WMMA_BF16(a1, b1, acc[1][1]);
    acc[0][2] = WMMA_BF16(a0, b2, acc[0][2]);
    acc[1][2] = WMMA_BF16(a1, b2, acc[1][2]);
    acc[0][3] = WMMA_BF16(a0, b3, acc[0][3]);
    acc[1][3] = WMMA_BF16(a1, b3, acc[1][3]);

    // C layout (ISA 7.12.2): VGPR v -> row v (+8 for lanes 16..31), col = lane&15
#pragma unroll
    for (int i = 0; i < 2; ++i) {
#pragma unroll
        for (int j = 0; j < 4; ++j) {
            int n = colBase + j * 16 + l16;
            if (n < N) {
                int r = rowBase + i * 16 + half * 8;
#pragma unroll
                for (int v = 0; v < 8; ++v)
                    C[(size_t)(r + v) * ldc + n] = acc[i][j][v];
            }
        }
    }
}

// ---------------------------------------------------------------------------
// Host-side orchestration
// ---------------------------------------------------------------------------
extern "C" void kernel_launch(void* const* d_in, const int* in_sizes, int n_in,
                              void* d_out, int out_size, void* d_ws, size_t ws_size,
                              hipStream_t stream)
{
    (void)in_sizes; (void)n_in; (void)out_size; (void)ws_size;

    const int*   tok    = (const int*)  d_in[0];
    const float* embed  = (const float*)d_in[1];
    const float* Wb     = (const float*)d_in[2];
    const float* n1g    = (const float*)d_in[3];
    const float* n1b    = (const float*)d_in[4];
    const float* n2g    = (const float*)d_in[5];
    const float* n2b    = (const float*)d_in[6];
    const float* normg  = (const float*)d_in[7];
    const float* normb  = (const float*)d_in[8];
    const float* outp   = (const float*)d_in[9];
    float* logits = (float*)d_out;

    // workspace carve-out (256B aligned)
    char* p = (char*)d_ws;
    auto carve = [&](size_t bytes) {
        void* r = (void*)p;
        p += (bytes + 255) & ~(size_t)255;
        return r;
    };
    float*          h    = (float*)         carve((size_t)T_SEQ * DIM_D  * 4);
    float*          y    = (float*)         carve((size_t)T_SEQ * DIM_D  * 4);
    float*          hn   = (float*)         carve((size_t)T_SEQ * DIM_D  * 4);
    __hip_bfloat16* hnb  = (__hip_bfloat16*)carve((size_t)T_SEQ * DIM_DP * 2);
    float*          Cbuf = (float*)         carve((size_t)T_SEQ * DIM_DP * 4);
    __hip_bfloat16* Wl   = (__hip_bfloat16*)carve((size_t)DIM_DP * DIM_DP * 2);
    __hip_bfloat16* OPb  = (__hip_bfloat16*)carve((size_t)VOCAB * DIM_DP * 2);

    // one-time conversions for this launch
    {
        size_t tot = (size_t)VOCAB * DIM_DP;
        convert_pad_bf16<<<(unsigned)((tot + 255) / 256), 256, 0, stream>>>(
            outp, OPb, VOCAB, DIM_D, VOCAB, DIM_DP);
    }
    {
        size_t tot = (size_t)T_SEQ * DIM_D;
        gather_embed<<<(unsigned)((tot + 255) / 256), 256, 0, stream>>>(tok, embed, h);
    }

    dim3 attnGrid((DIM_D + 255) / 256, T_SEQ / 128);
    dim3 gemmGridL(DIM_DP / 128, T_SEQ / 128);            // 15 x 32
    dim3 gemmGridF((VOCAB + 127) / 128, T_SEQ / 128);     // 79 x 32
    size_t wTot = (size_t)DIM_DP * DIM_DP;

    for (int l = 0; l < NLAYERS; ++l) {
        // attn + residual
        attn_resid<<<attnGrid, 256, 0, stream>>>(h, y);
        // hn = LN(y) ; also bf16-padded operand
        ln_fused<<<T_SEQ, 256, 0, stream>>>(y, nullptr, 1.0f, 0.0f,
                                            n1g + (size_t)l * DIM_D, n1b + (size_t)l * DIM_D,
                                            hn, hnb, 0);
        // this layer's W -> bf16 padded
        convert_pad_bf16<<<(unsigned)((wTot + 255) / 256), 256, 0, stream>>>(
            Wb + (size_t)l * DIM_D * DIM_D, Wl, DIM_D, DIM_D, DIM_DP, DIM_DP);
        // Cbuf = hn @ W^T
        wmma_gemm_nt_bf16<<<gemmGridL, 256, 0, stream>>>(hnb, Wl, Cbuf,
                                                         DIM_DP, DIM_DP, DIM_DP);
        // h = LN((1+alpha)*hn + beta*Cbuf)
        ln_fused<<<T_SEQ, 256, 0, stream>>>(hn, Cbuf, 1.0f + ALPHA_F, BETA_F,
                                            n2g + (size_t)l * DIM_D, n2b + (size_t)l * DIM_D,
                                            h, nullptr, DIM_DP);
    }

    // final LN -> bf16 operand, then logits = h @ out_proj^T
    ln_fused<<<T_SEQ, 256, 0, stream>>>(h, nullptr, 1.0f, 0.0f,
                                        normg, normb, nullptr, hnb, 0);
    wmma_gemm_nt_bf16<<<gemmGridF, 256, 0, stream>>>(hnb, OPb, logits,
                                                     VOCAB, DIM_DP, VOCAB);
}
